// SiglipEncoderLayer_4303557231410
// MI455X (gfx1250) — compile-verified, matrix operand order checked by reading, not run on hardware
//
#include <hip/hip_runtime.h>
#include <hip/hip_bf16.h>
#include <math.h>

typedef __attribute__((ext_vector_type(16))) _Float16 v16h;
typedef __attribute__((ext_vector_type(8)))  float    v8f;

#define WMMA_F16(a, b, c) \
  __builtin_amdgcn_wmma_f32_16x16x32_f16(false, (a), false, (b), (short)0, (c), false, false)

// ---------------------------------------------------------------------------
// Fragment load for 16x32 f16 A/B fragments (ISA 7.12.2 layout) from a
// per-lane pointer p = base + row*ld + kbase + (lane>>4)*8.
// halves 0..7  <- this lane's k-group, halves 8..15 <- k-group + 16.
// Two 16-byte loads per lane.
// ---------------------------------------------------------------------------
__device__ __forceinline__ v16h load_frag_p(const _Float16* p) {
  v16h f;
  *((uint4*)&f)     = *(const uint4*)(p);
  *((uint4*)&f + 1) = *(const uint4*)(p + 16);
  return f;
}

__device__ __forceinline__ float gelu_tanh(float x) {
  const float c = 0.7978845608028654f;   // sqrt(2/pi)
  return 0.5f * x * (1.0f + tanhf(c * (x + 0.044715f * x * x * x)));
}

// ---------------------------------------------------------------------------
// Weight convert: fp32 row-major [K][Nout] -> f16 row-major [Nout][K]
// ---------------------------------------------------------------------------
__global__ __launch_bounds__(256) void wcast_kernel(const float* __restrict__ w,
                                                    _Float16* __restrict__ wt,
                                                    int K, int Nout) {
  size_t idx = (size_t)blockIdx.x * blockDim.x + threadIdx.x;
  size_t total = (size_t)K * Nout;
  if (idx >= total) return;
  int k = (int)(idx / Nout);
  int n = (int)(idx % Nout);
  wt[(size_t)n * K + k] = (_Float16)w[idx];
}

// ---------------------------------------------------------------------------
// LayerNorm: fp32 [rows][E] -> f16 [rows][E]. One block per row.
// ---------------------------------------------------------------------------
__global__ __launch_bounds__(256) void ln_kernel(const float* __restrict__ x,
                                                 const float* __restrict__ g,
                                                 const float* __restrict__ b,
                                                 _Float16* __restrict__ out, int Ecols) {
  int row = blockIdx.x;
  int tid = threadIdx.x;
  const float* xr = x + (size_t)row * Ecols;
  __shared__ float red[256];

  float s = 0.f;
  for (int i = tid; i < Ecols; i += 256) s += xr[i];
  red[tid] = s; __syncthreads();
  for (int off = 128; off > 0; off >>= 1) {
    if (tid < off) red[tid] += red[tid + off];
    __syncthreads();
  }
  float mu = red[0] / Ecols;
  __syncthreads();

  float v = 0.f;
  for (int i = tid; i < Ecols; i += 256) { float d = xr[i] - mu; v += d * d; }
  red[tid] = v; __syncthreads();
  for (int off = 128; off > 0; off >>= 1) {
    if (tid < off) red[tid] += red[tid + off];
    __syncthreads();
  }
  float rstd = rsqrtf(red[0] / Ecols + 1e-6f);

  _Float16* outr = out + (size_t)row * Ecols;
  for (int i = tid; i < Ecols; i += 256)
    outr[i] = (_Float16)((xr[i] - mu) * rstd * g[i] + b[i]);
}

// ---------------------------------------------------------------------------
// V transpose: f16 [N][E] -> f16 [B,H,D,S]  (Vt[((b*H+h)*D+d)*S + s])
// ---------------------------------------------------------------------------
__global__ __launch_bounds__(256) void vtrans_kernel(const _Float16* __restrict__ v,
                                                     _Float16* __restrict__ vt,
                                                     int N, int S, int Hh, int Dd, int Ecols) {
  size_t idx = (size_t)blockIdx.x * blockDim.x + threadIdx.x;
  size_t total = (size_t)N * Ecols;
  if (idx >= total) return;
  int n = (int)(idx / Ecols);
  int c = (int)(idx % Ecols);
  int s = n % S, b = n / S;
  int h = c / Dd, d = c % Dd;
  vt[(((size_t)(b * Hh + h) * Dd) + d) * S + s] = v[idx];
}

// ---------------------------------------------------------------------------
// WMMA GEMM: out[N x Ncols] = A[N x K](f16) * Bt[Ncols x K](f16, pre-transposed)
//            + bias (+ optional GELU, residual). 1 wave -> 64x32 output tile
//            (4x2 grid of 16x16 WMMA frags -> 8 v_wmma per 32-K step).
// All fragment addresses derive directly from kernel args so addrspace
// inference yields global_load_b128 (not flat).
// ---------------------------------------------------------------------------
template <bool GELU, bool RESID, bool OUTF16>
__global__ __launch_bounds__(256) void gemm_kernel(
    const _Float16* __restrict__ A, const _Float16* __restrict__ Bt,
    const float* __restrict__ bias, const float* __restrict__ resid,
    float* __restrict__ outf, _Float16* __restrict__ outh,
    int Nrows, int Kdim, int Ncols) {
  int gwave = (int)((blockIdx.x * (size_t)blockDim.x + threadIdx.x) >> 5);
  int lane = threadIdx.x & 31;
  int tilesN = Ncols >> 5;   // 32-wide N tiles
  int tilesM = Nrows >> 6;   // 64-tall M tiles
  if (gwave >= tilesM * tilesN) return;
  int tm = (gwave / tilesN) << 6;
  int tn = (gwave % tilesN) << 5;
  int r = lane & 15, h = lane >> 4;

  const _Float16* Abase = A  + (size_t)(tm + r) * Kdim + h * 8;
  const _Float16* Bbase = Bt + (size_t)(tn + r) * Kdim + h * 8;
  const size_t rowStep = (size_t)16 * Kdim;

  v8f acc[4][2] = {};
  for (int k = 0; k < Kdim; k += 32) {
    v16h a[4], bf[2];
#pragma unroll
    for (int t = 0; t < 4; t++) a[t]  = load_frag_p(Abase + (size_t)t * rowStep + k);
#pragma unroll
    for (int t = 0; t < 2; t++) bf[t] = load_frag_p(Bbase + (size_t)t * rowStep + k);

    if (k + 64 < Kdim) {       // uniform: prefetch next-next K slice
#pragma unroll
      for (int t = 0; t < 4; t++) __builtin_prefetch(Abase + (size_t)t * rowStep + k + 64, 0, 1);
#pragma unroll
      for (int t = 0; t < 2; t++) __builtin_prefetch(Bbase + (size_t)t * rowStep + k + 64, 0, 1);
    }
#pragma unroll
    for (int i = 0; i < 4; i++)
#pragma unroll
      for (int j = 0; j < 2; j++)
        acc[i][j] = WMMA_F16(a[i], bf[j], acc[i][j]);
  }

#pragma unroll
  for (int i2 = 0; i2 < 4; i2++) {
#pragma unroll
    for (int j2 = 0; j2 < 2; j2++) {
      int rowb = tm + i2 * 16 + 8 * h;      // C/D layout: m = i + 8*h
      int colb = tn + j2 * 16 + r;          // n = lane (mod 16)
      float bv = bias[colb];
#pragma unroll
      for (int i = 0; i < 8; i++) {
        float val = acc[i2][j2][i] + bv;
        if (GELU) val = gelu_tanh(val);
        size_t off = (size_t)(rowb + i) * Ncols + colb;
        if (RESID) val += resid[off];
        if (OUTF16) outh[off] = (_Float16)val;
        else        outf[off] = val;
      }
    }
  }
}

// ---------------------------------------------------------------------------
// Flash attention: one wave (32-thread block) handles 32 query rows of one
// (b,h). Online softmax over 32-key blocks; every lane owns one softmax row.
// Two-pass softmax over the LDS score tile. QK^T and P*V via WMMA; V frags
// shared across both row tiles. Fragment addresses derive directly from
// kernel args (global loads, minimal loop-carried state).
//   Q, K : f16 [N][E]  (head slice at column h*D)
//   Vt   : f16 [B,H,D,S]
//   O    : f16 [N][E]
// ---------------------------------------------------------------------------
__global__ __launch_bounds__(32) void attn_kernel(
    const _Float16* __restrict__ Q, const _Float16* __restrict__ K,
    const _Float16* __restrict__ Vt, const float* __restrict__ mask,
    _Float16* __restrict__ O, int B, int S, int Hh, int Dd,
    int Ecols, float scale) {
  int qtiles = S >> 5;
  int qt = blockIdx.x % qtiles;
  int bh = blockIdx.x / qtiles;
  int hh = bh % Hh, b = bh / Hh;
  int lane = threadIdx.x;
  int r = lane & 15, h = lane >> 4;
  int q0 = qt << 5;

  __shared__ alignas(16) float    sS[32 * 32];   // score block, row-major [32][32]
  __shared__ alignas(16) _Float16 sP[32 * 32];   // exp(scores), f16
  __shared__ float sStat[32];                    // per-row alpha / 1/l broadcast

  // Q fragments (persistent)
  v16h qf[2][2];
#pragma unroll
  for (int rt = 0; rt < 2; rt++) {
    const _Float16* qp = Q + (size_t)(b * S + q0 + rt * 16 + r) * Ecols + hh * Dd + h * 8;
    qf[rt][0] = load_frag_p(qp);
    qf[rt][1] = load_frag_p(qp + 32);
  }

  // loop-invariant per-lane bases, direct from kernel args
  const _Float16* Kbase  = K  + (size_t)(b * S + r) * Ecols + hh * Dd + h * 8;
  const size_t    kRow16 = (size_t)16 * Ecols;
  const _Float16* Vbase  = Vt + ((size_t)(b * Hh + hh) * Dd + r) * S + h * 8;
  const size_t    vRow16 = (size_t)16 * S;
  const float*    maskRow = mask + (size_t)b * S * S + (size_t)(q0 + lane) * S;

  v8f o[2][4] = {};                    // 32 x 64 f32 accumulator
  float mrow = -INFINITY, lrow = 0.f;  // lane owns query row q0+lane

  for (int kt = 0; kt < S; kt += 32) {
    // --- scores for 32 keys x 32 queries ---
    {
      v16h kf[2][2];
#pragma unroll
      for (int sub = 0; sub < 2; sub++) {
        const _Float16* kp = Kbase + (size_t)(kt + sub * 16) * Ecols;
        kf[sub][0] = load_frag_p(kp);
        kf[sub][1] = load_frag_p(kp + 32);
      }
#pragma unroll
      for (int rt = 0; rt < 2; rt++) {
#pragma unroll
        for (int sub = 0; sub < 2; sub++) {
          v8f sc = {};
          sc = WMMA_F16(qf[rt][0], kf[sub][0], sc);
          sc = WMMA_F16(qf[rt][1], kf[sub][1], sc);
#pragma unroll
          for (int i = 0; i < 8; i++)
            sS[(rt * 16 + 8 * h + i) * 32 + sub * 16 + r] = sc[i];
        }
      }
    }
    __syncthreads();

    // --- online softmax, two passes over LDS: lane owns row q0+lane ---
    {
      const float* mrp = maskRow + kt;
      float bm = -INFINITY;
#pragma unroll
      for (int j = 0; j < 32; j++)
        bm = fmaxf(bm, sS[lane * 32 + j] * scale + mrp[j]);
      float nm = fmaxf(mrow, bm);
      float alpha = __expf(mrow - nm);
      float ls = 0.f;
#pragma unroll
      for (int j = 0; j < 32; j++) {
        float pv = __expf(sS[lane * 32 + j] * scale + mrp[j] - nm);
        ls += pv;
        sP[lane * 32 + j] = (_Float16)pv;
      }
      lrow = lrow * alpha + ls;
      mrow = nm;
      sStat[lane] = alpha;
    }
    __syncthreads();

    // --- rescale O, accumulate P @ V (V frags shared across row tiles) ---
    {
      float aL[2][8];
#pragma unroll
      for (int rt = 0; rt < 2; rt++)
#pragma unroll
        for (int i = 0; i < 8; i++) aL[rt][i] = sStat[rt * 16 + 8 * h + i];
#pragma unroll
      for (int rt = 0; rt < 2; rt++)
#pragma unroll
        for (int dt = 0; dt < 4; dt++)
#pragma unroll
          for (int i = 0; i < 8; i++) o[rt][dt][i] *= aL[rt][i];

      v16h pf0 = load_frag_p(sP + (size_t)r * 32 + h * 8);
      v16h pf1 = load_frag_p(sP + (size_t)(16 + r) * 32 + h * 8);
#pragma unroll
      for (int dt = 0; dt < 4; dt++) {
        v16h vf = load_frag_p(Vbase + (size_t)dt * vRow16 + kt);
        o[0][dt] = WMMA_F16(pf0, vf, o[0][dt]);
        o[1][dt] = WMMA_F16(pf1, vf, o[1][dt]);
      }
    }
    __syncthreads();
  }

  // --- normalize by l and store ---
  sStat[lane] = 1.0f / lrow;
  __syncthreads();
#pragma unroll
  for (int rt = 0; rt < 2; rt++) {
#pragma unroll
    for (int dt = 0; dt < 4; dt++) {
#pragma unroll
      for (int i = 0; i < 8; i++) {
        int m = rt * 16 + 8 * h + i;
        float inv = sStat[m];
        O[((size_t)(b * S + q0 + m)) * Ecols + hh * Dd + dt * 16 + r] =
            (_Float16)(o[rt][dt][i] * inv);
      }
    }
  }
}

// ---------------------------------------------------------------------------
// Host orchestration
// ---------------------------------------------------------------------------
extern "C" void kernel_launch(void* const* d_in, const int* in_sizes, int n_in,
                              void* d_out, int out_size, void* d_ws, size_t ws_size,
                              hipStream_t stream) {
  const int E = 768, H = 12, D = 64;
  const float* hidden = (const float*)d_in[0];
  const float* mask   = (const float*)d_in[1];
  const float* wq = (const float*)d_in[2];  const float* bq = (const float*)d_in[3];
  const float* wk = (const float*)d_in[4];  const float* bk = (const float*)d_in[5];
  const float* wv = (const float*)d_in[6];  const float* bv = (const float*)d_in[7];
  const float* wo = (const float*)d_in[8];  const float* bo = (const float*)d_in[9];
  const float* ln1_g = (const float*)d_in[10]; const float* ln1_b = (const float*)d_in[11];
  const float* ln2_g = (const float*)d_in[12]; const float* ln2_b = (const float*)d_in[13];
  const float* w1 = (const float*)d_in[14]; const float* b1 = (const float*)d_in[15];
  const float* w2 = (const float*)d_in[16]; const float* b2 = (const float*)d_in[17];
  const int I = in_sizes[15];               // 3072
  const int N = in_sizes[0] / E;            // B*S = 8192
  const int S = in_sizes[1] / N;            // 1024
  const int B = N / S;                      // 8
  float* out = (float*)d_out;

  char* p = (char*)d_ws;
  auto alloc = [&](size_t bytes) -> char* {
    char* r = p;
    p += (bytes + 255) & ~(size_t)255;
    return r;
  };
  _Float16* lnh  = (_Float16*)alloc((size_t)N * E * 2);
  _Float16* wqT  = (_Float16*)alloc((size_t)E * E * 2);
  _Float16* wkT  = (_Float16*)alloc((size_t)E * E * 2);
  _Float16* wvT  = (_Float16*)alloc((size_t)E * E * 2);
  _Float16* woT  = (_Float16*)alloc((size_t)E * E * 2);
  _Float16* w1T  = (_Float16*)alloc((size_t)E * I * 2);
  _Float16* w2T  = (_Float16*)alloc((size_t)I * E * 2);
  _Float16* Qh   = (_Float16*)alloc((size_t)N * E * 2);
  _Float16* Kh   = (_Float16*)alloc((size_t)N * E * 2);
  _Float16* Vh   = (_Float16*)alloc((size_t)N * E * 2);
  _Float16* VtT  = (_Float16*)alloc((size_t)N * E * 2);
  _Float16* Oh   = (_Float16*)alloc((size_t)N * E * 2);
  float*    xmid = (float*)   alloc((size_t)N * E * 4);
  _Float16* ln2h = (_Float16*)alloc((size_t)N * E * 2);
  _Float16* hmid = (_Float16*)alloc((size_t)N * I * 2);
  (void)ws_size; (void)n_in; (void)out_size;

  auto blks = [](size_t elems) { return (unsigned)((elems + 255) / 256); };

  // 1) weight transpose+cast
  wcast_kernel<<<blks((size_t)E * E), 256, 0, stream>>>(wq, wqT, E, E);
  wcast_kernel<<<blks((size_t)E * E), 256, 0, stream>>>(wk, wkT, E, E);
  wcast_kernel<<<blks((size_t)E * E), 256, 0, stream>>>(wv, wvT, E, E);
  wcast_kernel<<<blks((size_t)E * E), 256, 0, stream>>>(wo, woT, E, E);
  wcast_kernel<<<blks((size_t)E * I), 256, 0, stream>>>(w1, w1T, E, I);
  wcast_kernel<<<blks((size_t)I * E), 256, 0, stream>>>(w2, w2T, I, E);

  // 2) LN1
  ln_kernel<<<N, 256, 0, stream>>>(hidden, ln1_g, ln1_b, lnh, E);

  // 3) Q/K/V projections (f16 out)
  {
    int tiles = (N / 64) * (E / 32);
    unsigned nb = (unsigned)((tiles + 7) / 8);
    gemm_kernel<false, false, true><<<nb, 256, 0, stream>>>(lnh, wqT, bq, nullptr, nullptr, Qh, N, E, E);
    gemm_kernel<false, false, true><<<nb, 256, 0, stream>>>(lnh, wkT, bk, nullptr, nullptr, Kh, N, E, E);
    gemm_kernel<false, false, true><<<nb, 256, 0, stream>>>(lnh, wvT, bv, nullptr, nullptr, Vh, N, E, E);
  }

  // 4) V transpose -> [B,H,D,S]
  vtrans_kernel<<<blks((size_t)N * E), 256, 0, stream>>>(Vh, VtT, N, S, H, D, E);

  // 5) attention (32 query rows per wave)
  {
    float scale = rsqrtf((float)D);
    unsigned nb = (unsigned)(B * H * (S / 32));
    attn_kernel<<<nb, 32, 0, stream>>>(Qh, Kh, VtT, mask, Oh, B, S, H, D, E, scale);
  }

  // 6) output projection + residual -> xmid (fp32)
  {
    int tiles = (N / 64) * (E / 32);
    unsigned nb = (unsigned)((tiles + 7) / 8);
    gemm_kernel<false, true, false><<<nb, 256, 0, stream>>>(Oh, woT, bo, hidden, xmid, nullptr, N, E, E);
  }

  // 7) LN2
  ln_kernel<<<N, 256, 0, stream>>>(xmid, ln2_g, ln2_b, ln2h, E);

  // 8) MLP up + GELU (f16 out)
  {
    int tiles = (N / 64) * (I / 32);
    unsigned nb = (unsigned)((tiles + 7) / 8);
    gemm_kernel<true, false, true><<<nb, 256, 0, stream>>>(ln2h, w1T, b1, nullptr, nullptr, hmid, N, E, I);
  }

  // 9) MLP down + residual -> out (fp32)
  {
    int tiles = (N / 64) * (E / 32);
    unsigned nb = (unsigned)((tiles + 7) / 8);
    gemm_kernel<false, true, false><<<nb, 256, 0, stream>>>(hmid, w2T, b2, xmid, out, nullptr, N, I, E);
  }
}
